// SelfAttention_34050500722907
// MI455X (gfx1250) — matrix operand order repaired and checked by reading, not compiled, not run on hardware
//
#include <hip/hip_runtime.h>
#include <hip/hip_bf16.h>
#include <math.h>

// ---------------------------------------------------------------------------
// CDNA5 (gfx1250, wave32) WMMA types + helpers
// ---------------------------------------------------------------------------
typedef __attribute__((ext_vector_type(16))) __bf16 v16bf;
typedef __attribute__((ext_vector_type(8)))  float  v8f;
typedef int v4i_vec __attribute__((vector_size(16)));

union Frag {
  v16bf v;
  uint4 q[2];
};

__device__ __forceinline__ unsigned short f2bf(float f) {
  unsigned int u = __float_as_uint(f);
  u += 0x7FFFu + ((u >> 16) & 1u);   // round-to-nearest-even
  return (unsigned short)(u >> 16);
}
__device__ __forceinline__ float bf2f(unsigned short u) {
  return __uint_as_float(((unsigned int)u) << 16);
}

__device__ __forceinline__ v8f wmma_bf16(const Frag& a, const Frag& b, v8f c) {
  return __builtin_amdgcn_wmma_f32_16x16x32_bf16(
      false, a.v, false, b.v, (short)0, c, false, false);
}

// A-fragment (16x32 bf16): lane l (row = l&15, g = l>>4):
//   VGPR0-3: K = kbase + g*8 + [0..7], VGPR4-7: K = kbase + 16 + g*8 + [0..7]
__device__ __forceinline__ void load_a(Frag& f, const unsigned short* row, int kbase, int g) {
  f.q[0] = *(const uint4*)(row + kbase + g * 8);
  f.q[1] = *(const uint4*)(row + kbase + 16 + g * 8);
}
// B-fragment (32x16 bf16): lane l (col = l&15, g = l>>4):
//   VGPR0-3: K = kbase + g*8 + [0..7], VGPR4-7: K = kbase + 16 + g*8 + [0..7]
__device__ __forceinline__ void load_b(Frag& f, const unsigned short* col, int kbase, int g) {
  f.q[0] = *(const uint4*)(col + kbase + g * 8);
  f.q[1] = *(const uint4*)(col + kbase + 16 + g * 8);
}

// ---------------------------------------------------------------------------
// CDNA5 async global->LDS copy (ASYNCcnt path), with safe fallback
// ---------------------------------------------------------------------------
#define AS1 __attribute__((address_space(1)))
#define AS3 __attribute__((address_space(3)))

#if __has_builtin(__builtin_amdgcn_global_load_async_to_lds_b128)
#define HAVE_ASYNC_LDS 1
#endif

__device__ __forceinline__ void cp_b128(const unsigned short* g, unsigned short* l) {
#if defined(HAVE_ASYNC_LDS)
  __builtin_amdgcn_global_load_async_to_lds_b128(
      (AS1 v4i_vec*)g, (AS3 v4i_vec*)l, 0, 0);
#else
  *(uint4*)l = *(const uint4*)g;
#endif
}

__device__ __forceinline__ void wait_async() {
#if defined(HAVE_ASYNC_LDS)
#if __has_builtin(__builtin_amdgcn_s_wait_asynccnt)
  __builtin_amdgcn_s_wait_asynccnt(0);
#else
  asm volatile("s_wait_asynccnt 0x0" ::: "memory");
#endif
#endif
}

// ---------------------------------------------------------------------------
// f32 -> bf16 flat convert (8 elems/thread)
// ---------------------------------------------------------------------------
__global__ void f32_to_bf16_kernel(const float* __restrict__ in,
                                   unsigned short* __restrict__ out, size_t n8) {
  size_t i = ((size_t)blockIdx.x * blockDim.x + threadIdx.x);
  if (i >= n8) return;
  i *= 8;
  float4 a = *(const float4*)(in + i);
  float4 b = *(const float4*)(in + i + 4);
  unsigned short t[8] = {f2bf(a.x), f2bf(a.y), f2bf(a.z), f2bf(a.w),
                         f2bf(b.x), f2bf(b.y), f2bf(b.z), f2bf(b.w)};
  *(uint4*)(out + i) = *(uint4*)t;
}

// ---------------------------------------------------------------------------
// W [K][N] f32 -> Wt [N][K] bf16 (64x64 tiles via LDS)
// ---------------------------------------------------------------------------
__global__ void __launch_bounds__(256)
transpose_w_kernel(const float* __restrict__ W, unsigned short* __restrict__ Wt,
                   int K, int N) {
  __shared__ __align__(16) unsigned short Ls[64 * 72];   // [n][k]
  const int n0 = blockIdx.x * 64, k0 = blockIdx.y * 64;
  const int tid = threadIdx.x;
  {
    int kr = tid >> 2;
    int nq = (tid & 3) * 16;
    const float4* src = (const float4*)(W + (size_t)(k0 + kr) * N + n0 + nq);
    #pragma unroll
    for (int j = 0; j < 4; ++j) {
      float4 v = src[j];
      int nb = nq + j * 4;
      Ls[(nb + 0) * 72 + kr] = f2bf(v.x);
      Ls[(nb + 1) * 72 + kr] = f2bf(v.y);
      Ls[(nb + 2) * 72 + kr] = f2bf(v.z);
      Ls[(nb + 3) * 72 + kr] = f2bf(v.w);
    }
  }
  __syncthreads();
  {
    int nr = tid >> 2;
    int kq = (tid & 3) * 16;
    unsigned short tmp[16];
    #pragma unroll
    for (int j = 0; j < 16; ++j) tmp[j] = Ls[nr * 72 + kq + j];
    unsigned short* dst = Wt + (size_t)(n0 + nr) * K + k0 + kq;
    *(uint4*)dst = *(uint4*)&tmp[0];
    *(uint4*)(dst + 8) = *(uint4*)&tmp[8];
  }
}

// ---------------------------------------------------------------------------
// V slice of bf16 qkv -> Vt[(b*H+h)*64 + d][t] bf16 (64x64 tiles)
// ---------------------------------------------------------------------------
__global__ void __launch_bounds__(256)
transpose_v_kernel(const unsigned short* __restrict__ qkv16,
                   unsigned short* __restrict__ Vt, int T, int C, int H) {
  __shared__ __align__(16) unsigned short Ls[64 * 72];   // [d][t]
  const int tb = blockIdx.x, h = blockIdx.y, b = blockIdx.z;
  const int tid = threadIdx.x;
  const int row3C = 3 * C;
  {
    int t = tid >> 2;
    int dq = (tid & 3) * 16;
    const uint4* src =
        (const uint4*)(qkv16 + (size_t)(b * T + tb * 64 + t) * row3C + 2 * C + h * 64 + dq);
    unsigned short tmp[16];
    *(uint4*)&tmp[0] = src[0];
    *(uint4*)&tmp[8] = src[1];
    #pragma unroll
    for (int j = 0; j < 16; ++j) Ls[(dq + j) * 72 + t] = tmp[j];
  }
  __syncthreads();
  {
    int d = tid >> 2;
    int tq = (tid & 3) * 16;
    unsigned short tmp[16];
    #pragma unroll
    for (int j = 0; j < 16; ++j) tmp[j] = Ls[d * 72 + tq + j];
    unsigned short* dst = Vt + ((size_t)((b * H + h) * 64 + d)) * T + tb * 64 + tq;
    *(uint4*)dst = *(uint4*)&tmp[0];
    *(uint4*)(dst + 8) = *(uint4*)&tmp[8];
  }
}

// ---------------------------------------------------------------------------
// ||row||^2 of a bf16 matrix (used for x-norms and, on Wt, the w-norms)
// ---------------------------------------------------------------------------
__global__ void rownorm_bf16_kernel(const unsigned short* __restrict__ X,
                                    float* __restrict__ out, int Cdim) {
  int row = blockIdx.x;
  const unsigned short* p = X + (size_t)row * Cdim;
  float s = 0.f;
  for (int i = threadIdx.x; i < Cdim; i += blockDim.x) {
    float v = bf2f(p[i]);
    s += v * v;
  }
  for (int off = 16; off > 0; off >>= 1) s += __shfl_xor(s, off, 32);
  __shared__ float red[8];
  if ((threadIdx.x & 31) == 0) red[threadIdx.x >> 5] = s;
  __syncthreads();
  if (threadIdx.x == 0) {
    float t = 0.f;
    int nw = (blockDim.x + 31) >> 5;
    for (int i = 0; i < nw; ++i) t += red[i];
    out[row] = t;
  }
}

// ---------------------------------------------------------------------------
// scale = (sqrt(F)/log1p(F))^alpha, computed once (keeps powf out of GEMM)
// ---------------------------------------------------------------------------
__global__ void prep_scales_kernel(const float* aA, const float* aP, float* sc,
                                   float F1, float F2) {
  if (threadIdx.x == 0 && blockIdx.x == 0) {
    sc[0] = powf(sqrtf(F1) / log1pf(F1), aA[0]);
    sc[1] = powf(sqrtf(F2) / log1pf(F2), aP[0]);
  }
}

// ---------------------------------------------------------------------------
// yat_dense GEMM, bf16 in (A [M][K], Bt [N][K]), double-buffered async tiles.
// Block tile 128x64, BK=64, 256 threads = 8 waves (4x2), wave tile 32x32.
// ---------------------------------------------------------------------------
template <bool OUT_BF16>
__global__ void __launch_bounds__(256)
yat_gemm_kernel(const unsigned short* __restrict__ A, const unsigned short* __restrict__ Bt,
                const float* __restrict__ bias, const float* __restrict__ xnorm,
                const float* __restrict__ wnorm, const float* __restrict__ scale_p,
                void* __restrict__ Ov, int M, int N, int K) {
  __shared__ __align__(16) unsigned short As[2][128 * 72];
  __shared__ __align__(16) unsigned short Bs[2][64 * 72];

  const int tid = threadIdx.x;
  const int wid = tid >> 5;
  const int l   = tid & 31;
  const int g   = l >> 4;
  const int ln  = l & 15;
  const int wm  = wid >> 1;
  const int wn  = wid & 1;
  const int mblk = blockIdx.y * 128;
  const int nblk = blockIdx.x * 64;

  auto fill = [&](int buf, int kb) {
    {  // A tile 128x64: 4 x b128 per thread
      int r = tid >> 1, half = (tid & 1) * 32;
      const unsigned short* gp = A + (size_t)(mblk + r) * K + kb + half;
      unsigned short* lp = &As[buf][r * 72 + half];
      #pragma unroll
      for (int j = 0; j < 4; ++j) cp_b128(gp + j * 8, lp + j * 8);
    }
    {  // B tile 64x64: 2 x b128 per thread
      int n = tid >> 2, q = (tid & 3) * 16;
      const unsigned short* gp = Bt + (size_t)(nblk + n) * K + kb + q;
      unsigned short* lp = &Bs[buf][n * 72 + q];
      #pragma unroll
      for (int j = 0; j < 2; ++j) cp_b128(gp + j * 8, lp + j * 8);
    }
  };

  const v8f zero = {0.f, 0.f, 0.f, 0.f, 0.f, 0.f, 0.f, 0.f};
  v8f c00 = zero, c01 = zero, c10 = zero, c11 = zero;

  int cur = 0;
  fill(0, 0);
  wait_async();
  __syncthreads();

  for (int kb = 0; kb < K; kb += 64) {
    if (kb + 64 < K) fill(cur ^ 1, kb + 64);
    #pragma unroll
    for (int kc = 0; kc < 2; ++kc) {
      Frag a0, a1, b0, b1;
      load_a(a0, &As[cur][(wm * 32 +      ln) * 72], kc * 32, g);
      load_a(a1, &As[cur][(wm * 32 + 16 + ln) * 72], kc * 32, g);
      load_b(b0, &Bs[cur][(wn * 32 +      ln) * 72], kc * 32, g);
      load_b(b1, &Bs[cur][(wn * 32 + 16 + ln) * 72], kc * 32, g);
      c00 = wmma_bf16(a0, b0, c00);
      c01 = wmma_bf16(a0, b1, c01);
      c10 = wmma_bf16(a1, b0, c10);
      c11 = wmma_bf16(a1, b1, c11);
    }
    wait_async();
    __syncthreads();
    cur ^= 1;
  }

  const float scale = scale_p[0];
  v8f acc[2][2] = {{c00, c01}, {c10, c11}};
  #pragma unroll
  for (int mt = 0; mt < 2; ++mt) {
    #pragma unroll
    for (int nt = 0; nt < 2; ++nt) {
      int n = nblk + wn * 32 + nt * 16 + ln;
      float wnv = wnorm[n];
      float bv  = bias[n];
      int mbase = mblk + wm * 32 + mt * 16 + g * 8;
      #pragma unroll
      for (int r = 0; r < 8; ++r) {
        float y  = acc[mt][nt][r];
        float xn = xnorm[mbase + r];
        float d  = xn + wnv - 2.f * y + 1e-5f;
        float o  = (y * y / d) * scale + bv;
        size_t idx = (size_t)(mbase + r) * N + n;
        if (OUT_BF16) ((unsigned short*)Ov)[idx] = f2bf(o);
        else          ((float*)Ov)[idx] = o;
      }
    }
  }
}

// ---------------------------------------------------------------------------
// Flash attention (causal, H=16, hd=64), bf16 qkv + pre-transposed Vt.
// Block = 64 q-rows of one (b,h); 4 waves x 16 q-rows; K/V double-buffered.
// ---------------------------------------------------------------------------
__global__ void __launch_bounds__(128)
flash_attn_kernel(const unsigned short* __restrict__ qkv16,
                  const unsigned short* __restrict__ Vt16,
                  unsigned short* __restrict__ out16,
                  int B, int T, int C, int H) {
  const int qb  = blockIdx.x;
  const int h   = blockIdx.y;
  const int b   = blockIdx.z;
  const int tid = threadIdx.x;
  const int wid = tid >> 5;
  const int l   = tid & 31;
  const int g   = l >> 4;
  const int ln  = l & 15;
  const int row3C = 3 * C;

  __shared__ __align__(16) unsigned short Qs[64 * 72];      // [q][d]
  __shared__ __align__(16) unsigned short Ks[2][64 * 72];   // [k][d]
  __shared__ __align__(16) unsigned short Vts[2][64 * 72];  // [d][k]
  __shared__ __align__(16) unsigned short Ps[64 * 72];      // [q][k] wave-private rows

  auto fillKV = [&](int buf, int kb) {
    int r = tid >> 1, half = (tid & 1) * 32;
    const unsigned short* kg =
        qkv16 + (size_t)(b * T + kb * 64 + r) * row3C + C + h * 64 + half;
    unsigned short* kl = &Ks[buf][r * 72 + half];
    #pragma unroll
    for (int j = 0; j < 4; ++j) cp_b128(kg + j * 8, kl + j * 8);
    const unsigned short* vg =
        Vt16 + ((size_t)((b * H + h) * 64 + r)) * T + kb * 64 + half;
    unsigned short* vl = &Vts[buf][r * 72 + half];
    #pragma unroll
    for (int j = 0; j < 4; ++j) cp_b128(vg + j * 8, vl + j * 8);
  };

  {  // Q block (once)
    int r = tid >> 1, half = (tid & 1) * 32;
    const unsigned short* qg =
        qkv16 + (size_t)(b * T + qb * 64 + r) * row3C + h * 64 + half;
    unsigned short* ql = &Qs[r * 72 + half];
    #pragma unroll
    for (int j = 0; j < 4; ++j) cp_b128(qg + j * 8, ql + j * 8);
  }
  fillKV(0, 0);
  wait_async();
  __syncthreads();

  Frag aq0, aq1;
  {
    const unsigned short* qrow = Qs + (wid * 16 + ln) * 72;
    load_a(aq0, qrow, 0,  g);
    load_a(aq1, qrow, 32, g);
  }

  const v8f zero = {0.f, 0.f, 0.f, 0.f, 0.f, 0.f, 0.f, 0.f};
  v8f o[4];
  o[0] = zero; o[1] = zero; o[2] = zero; o[3] = zero;
  float m_run[8], l_run[8];
  #pragma unroll
  for (int r = 0; r < 8; ++r) { m_run[r] = -3.0e38f; l_run[r] = 0.f; }
  const int qrow_g = qb * 64 + wid * 16 + g * 8;

  int cur = 0;
  for (int kb = 0; kb <= qb; ++kb) {
    if (kb + 1 <= qb) fillKV(cur ^ 1, kb + 1);

    // ---- S = Q K^T (16 x 64 per wave), then apply 1/sqrt(hd) ----
    v8f s[4];
    #pragma unroll
    for (int nt = 0; nt < 4; ++nt) {
      Frag bk0, bk1;
      const unsigned short* kcol = &Ks[cur][(nt * 16 + ln) * 72];
      load_b(bk0, kcol, 0,  g);
      load_b(bk1, kcol, 32, g);
      s[nt] = wmma_bf16(aq0, bk0, zero);
      s[nt] = wmma_bf16(aq1, bk1, s[nt]);
      #pragma unroll
      for (int r = 0; r < 8; ++r) s[nt][r] *= 0.125f;
    }

    if (kb == qb) {  // causal mask on diagonal block
      #pragma unroll
      for (int nt = 0; nt < 4; ++nt) {
        int kg = kb * 64 + nt * 16 + ln;
        #pragma unroll
        for (int r = 0; r < 8; ++r)
          if (kg > qrow_g + r) s[nt][r] = -3.0e38f;
      }
    }

    // ---- online softmax ----
    float mx[8];
    #pragma unroll
    for (int r = 0; r < 8; ++r)
      mx[r] = fmaxf(fmaxf(s[0][r], s[1][r]), fmaxf(s[2][r], s[3][r]));
    #pragma unroll
    for (int off = 1; off < 16; off <<= 1)
      #pragma unroll
      for (int r = 0; r < 8; ++r)
        mx[r] = fmaxf(mx[r], __shfl_xor(mx[r], off, 32));

    float corr[8];
    #pragma unroll
    for (int r = 0; r < 8; ++r) {
      float mn = fmaxf(m_run[r], mx[r]);
      corr[r]  = __expf(m_run[r] - mn);
      m_run[r] = mn;
      l_run[r] *= corr[r];
    }
    #pragma unroll
    for (int nt = 0; nt < 4; ++nt)
      #pragma unroll
      for (int r = 0; r < 8; ++r)
        o[nt][r] *= corr[r];

    float rs[8];
    #pragma unroll
    for (int r = 0; r < 8; ++r) rs[r] = 0.f;
    #pragma unroll
    for (int nt = 0; nt < 4; ++nt)
      #pragma unroll
      for (int r = 0; r < 8; ++r) {
        float p = __expf(s[nt][r] - m_run[r]);
        s[nt][r] = p;
        rs[r] += p;
      }
    #pragma unroll
    for (int off = 1; off < 16; off <<= 1)
      #pragma unroll
      for (int r = 0; r < 8; ++r)
        rs[r] += __shfl_xor(rs[r], off, 32);
    #pragma unroll
    for (int r = 0; r < 8; ++r) l_run[r] += rs[r];

    // ---- stage P to LDS (A-fragment layout re-entry) ----
    #pragma unroll
    for (int nt = 0; nt < 4; ++nt)
      #pragma unroll
      for (int r = 0; r < 8; ++r)
        Ps[(wid * 16 + g * 8 + r) * 72 + nt * 16 + ln] = f2bf(s[nt][r]);

    // ---- O += P V ----
    Frag ap0, ap1;
    const unsigned short* prow = Ps + (wid * 16 + ln) * 72;
    load_a(ap0, prow, 0,  g);
    load_a(ap1, prow, 32, g);
    #pragma unroll
    for (int nt = 0; nt < 4; ++nt) {
      Frag bv0, bv1;
      const unsigned short* vcol = &Vts[cur][(nt * 16 + ln) * 72];
      load_b(bv0, vcol, 0,  g);
      load_b(bv1, vcol, 32, g);
      o[nt] = wmma_bf16(ap0, bv0, o[nt]);
      o[nt] = wmma_bf16(ap1, bv1, o[nt]);
    }

    wait_async();
    __syncthreads();
    cur ^= 1;
  }

  // ---- normalize + write bf16 attention output [B,T,C] ----
  #pragma unroll
  for (int nt = 0; nt < 4; ++nt)
    #pragma unroll
    for (int r = 0; r < 8; ++r) {
      int qg = qrow_g + r;
      out16[(size_t)(b * T + qg) * C + h * 64 + nt * 16 + ln] = f2bf(o[nt][r] / l_run[r]);
    }
}

// ---------------------------------------------------------------------------
// Host launch
// ---------------------------------------------------------------------------
extern "C" void kernel_launch(void* const* d_in, const int* in_sizes, int n_in,
                              void* d_out, int out_size, void* d_ws, size_t ws_size,
                              hipStream_t stream) {
  (void)in_sizes; (void)n_in; (void)out_size; (void)ws_size;
  const float* x          = (const float*)d_in[0];
  /* d_in[1] = causal mask (tril) — realized directly in flash_attn_kernel */
  const float* W_attn     = (const float*)d_in[2];
  const float* b_attn     = (const float*)d_in[3];
  const float* alpha_attn = (const float*)d_in[4];
  const float* W_proj     = (const float*)d_in[5];
  const float* b_proj     = (const float*)d_in[6];
  const float* alpha_proj = (const float*)d_in[7];
  float* out = (float*)d_out;

  const int B = 8, T = 1024, C = 1024, H = 16;
  const int M = B * T;        // 8192
  const int N1 = 3 * C;       // 3072
  const int K = C;

  // workspace layout (bytes, 256B aligned chunks)
  char* ws = (char*)d_ws;
  size_t off = 0;
  auto alloc = [&](size_t bytes) {
    void* p = ws + off;
    off += (bytes + 255) & ~(size_t)255;
    return p;
  };
  unsigned short* qkv16 = (unsigned short*)alloc((size_t)M * N1 * 2);  // 50.3 MB
  unsigned short* att16 = (unsigned short*)alloc((size_t)M * C * 2);   // 16.8 MB
  unsigned short* Vt16  = (unsigned short*)alloc((size_t)M * C * 2);   // 16.8 MB
  unsigned short* xbf   = (unsigned short*)alloc((size_t)M * K * 2);   // 16.8 MB
  unsigned short* WtA   = (unsigned short*)alloc((size_t)N1 * K * 2);  //  6.3 MB
  unsigned short* WtP   = (unsigned short*)alloc((size_t)C * K * 2);   //  2.1 MB
  float* xn1 = (float*)alloc((size_t)M * 4);
  float* xn2 = (float*)alloc((size_t)M * 4);
  float* wnA = (float*)alloc((size_t)N1 * 4);
  float* wnP = (float*)alloc((size_t)C * 4);
  float* sc  = (float*)alloc(2 * 4);

  // --- operand preparation ---
  {
    size_t n8 = (size_t)M * K / 8;
    f32_to_bf16_kernel<<<(unsigned)((n8 + 255) / 256), 256, 0, stream>>>(x, xbf, n8);
  }
  transpose_w_kernel<<<dim3(N1 / 64, K / 64), 256, 0, stream>>>(W_attn, WtA, K, N1);
  transpose_w_kernel<<<dim3(C / 64, K / 64), 256, 0, stream>>>(W_proj, WtP, K, C);
  prep_scales_kernel<<<1, 32, 0, stream>>>(alpha_attn, alpha_proj, sc, (float)N1, (float)C);
  rownorm_bf16_kernel<<<M, 256, 0, stream>>>(xbf, xn1, K);
  rownorm_bf16_kernel<<<N1, 256, 0, stream>>>(WtA, wnA, K);
  rownorm_bf16_kernel<<<C, 256, 0, stream>>>(WtP, wnP, K);

  // --- qkv = yat_dense(x, W_attn) -> bf16 ---
  yat_gemm_kernel<true><<<dim3(N1 / 64, M / 128), 256, 0, stream>>>(
      xbf, WtA, b_attn, xn1, wnA, sc + 0, qkv16, M, N1, K);

  // --- attention ---
  transpose_v_kernel<<<dim3(T / 64, H, B), 256, 0, stream>>>(qkv16, Vt16, T, C, H);
  flash_attn_kernel<<<dim3(T / 64, H, B), 128, 0, stream>>>(qkv16, Vt16, att16, B, T, C, H);

  // --- out = yat_dense(att, W_proj) -> f32 ---
  rownorm_bf16_kernel<<<M, 256, 0, stream>>>(att16, xn2, K);
  yat_gemm_kernel<false><<<dim3(C / 64, M / 128), 256, 0, stream>>>(
      att16, WtP, b_proj, xn2, wnP, sc + 1, out, M, C, C);
}
